// ORFAttn_56530359550030
// MI455X (gfx1250) — matrix operand order repaired and checked by reading, not compiled, not run on hardware
//
#include <hip/hip_runtime.h>
#include <hip/hip_bf16.h>
#include <math.h>

// ---- CDNA5 WMMA types -------------------------------------------------------
typedef _Float16 v16h  __attribute__((ext_vector_type(16)));
typedef _Float16 half8 __attribute__((ext_vector_type(8)));
typedef float    v8f   __attribute__((ext_vector_type(8)));

#define DEVINL static __device__ __forceinline__

DEVINL v8f wmma_f16(v16h a, v16h b, v8f c) {
  // D(16x16,f32) = A(16x32,f16) x B(32x16,f16) + C
  return __builtin_amdgcn_wmma_f32_16x16x32_f16(false, a, false, b, (short)0, c,
                                                false, false);
}

// A-matrix lane fragment: row M = lane%16, K halves = 16*hi + {0..15} (one 32B chunk)
// p = row base + 32*kchunk (halves). The 32B chunk is 32B aligned by construction.
DEVINL v16h ldsA(const _Float16* p, int hi) {
  return *(const v16h*)(p + 16 * hi);
}
// B-matrix lane fragment: col N = lane%16, K halves = 16*hi + {0..15} (one 32B chunk).
// p = col base + 32*kchunk (halves); source stored [N][K] row-major, 32B aligned rows.
DEVINL v16h ldsB(const _Float16* p, int hi) {
  return *(const v16h*)(p + 16 * hi);
}

#define INV_SIGMA 0.35355339059327373f /* 64^-0.25 */
#define FINV      0.0625f              /* 256^-0.5  */

// ---- K1: mean over S of k, per (b,d) ---------------------------------------
__global__ __launch_bounds__(256) void orf_mean_k(const float* __restrict__ kk,
                                                  float* __restrict__ meanw) {
  __shared__ float red[4][64];
  int b = blockIdx.x;
  int d = threadIdx.x & 63;
  int rg = threadIdx.x >> 6;
  const float* kb = kk + (size_t)b * 4096 * 64;
  float s = 0.f;
  for (int srow = rg; srow < 4096; srow += 4) s += kb[srow * 64 + d];
  red[rg][d] = s;
  __syncthreads();
  if (threadIdx.x < 64) {
    float m = (red[0][threadIdx.x] + red[1][threadIdx.x] +
               red[2][threadIdx.x] + red[3][threadIdx.x]) * (1.0f / 4096.0f);
    meanw[b * 64 + threadIdx.x] = m;
  }
}

// ---- K2: per-block max of ((k-mean)/sigma) @ W^T  (WMMA) -------------------
__global__ __launch_bounds__(256) void orf_kproj_max(const float* __restrict__ kk,
                                                     const float* __restrict__ wgt,
                                                     const float* __restrict__ meanw,
                                                     float* __restrict__ bmax) {
  extern __shared__ __align__(64) char smem[];
  _Float16* Wl    = (_Float16*)smem;             // 256*64 f16  (32768 B)
  _Float16* kcl   = (_Float16*)(smem + 32768);   // 128*64 f16  (16384 B)
  float*    meanl = (float*)(smem + 49152);      // 64 f32
  float*    wred  = (float*)(smem + 49408);      // 8 f32
  int b = blockIdx.y, sc = blockIdx.x, tid = threadIdx.x;

  for (int i = tid; i < 256 * 64; i += 256) Wl[i] = (_Float16)wgt[i];
  if (tid < 64) meanl[tid] = meanw[b * 64 + tid];
  __syncthreads();
  const float* kb = kk + ((size_t)b * 4096 + (size_t)sc * 128) * 64;
  for (int i = tid; i < 128 * 64; i += 256) {
    int d = i & 63;
    kcl[i] = (_Float16)((kb[i] - meanl[d]) * INV_SIGMA);
  }
  __syncthreads();

  int lane = tid & 31, w = tid >> 5, hi = lane >> 4, ln = lane & 15;
  float lm = -1e30f;
  for (int t = 0; t < 16; ++t) {
    v8f acc = {};
    for (int c = 0; c < 2; ++c) {
      v16h a  = ldsA(kcl + (16 * w + ln) * 64 + 32 * c, hi);
      v16h bm = ldsB(Wl + (16 * t + ln) * 64 + 32 * c, hi);
      acc = wmma_f16(a, bm, acc);
    }
#pragma unroll
    for (int r = 0; r < 8; ++r) lm = fmaxf(lm, acc[r]);
  }
#pragma unroll
  for (int m = 1; m <= 16; m <<= 1) lm = fmaxf(lm, __shfl_xor(lm, m));
  if (lane == 0) wred[w] = lm;
  __syncthreads();
  if (tid == 0) {
    float m = wred[0];
    for (int i = 1; i < 8; ++i) m = fmaxf(m, wred[i]);
    bmax[b * 32 + sc] = m;
  }
}

// ---- K3: reduce 32 block maxima -> Mk[b] -----------------------------------
__global__ void orf_max_reduce(const float* __restrict__ bmax, float* __restrict__ Mk) {
  int b = blockIdx.x, lane = threadIdx.x;
  float m = bmax[b * 32 + lane];
#pragma unroll
  for (int s = 1; s <= 16; s <<= 1) m = fmaxf(m, __shfl_xor(m, s));
  if (lane == 0) Mk[b] = m;
}

// ---- K4: phi_k, kv = phi_k^T @ [V | 1], ksum  (block = one (b, 16-f strip)) -
__global__ __launch_bounds__(256) void orf_phik_kv(const float* __restrict__ kk,
                                                   const float* __restrict__ vv,
                                                   const float* __restrict__ wgt,
                                                   const float* __restrict__ meanw,
                                                   const float* __restrict__ Mk,
                                                   float* __restrict__ kvw,
                                                   float* __restrict__ ksumw) {
  extern __shared__ __align__(64) char smem[];
  _Float16* Wl    = (_Float16*)smem;             // 16*64 f16       @0      (2048)
  _Float16* kcl   = (_Float16*)(smem + 2048);    // 256*64 f16              (32768)
  _Float16* vtl   = (_Float16*)(smem + 34816);   // 80 x 272 f16 (vT+ones)  (43520)
  float*    norml = (float*)(smem + 78336);      // 256 f32                 (1024)
  float*    meanl = (float*)(smem + 79360);      // 64 f32                  (256)
  _Float16* phiT  = (_Float16*)(smem + 79616);   // 8 waves x 16 x 32 f16   (8192)
  float*    kvred = (float*)(smem + 87808);      // 8 x 16 x 80 f32         (40960)
  int b = blockIdx.y, fi = blockIdx.x, tid = threadIdx.x;
  int f0 = fi * 16;

  for (int i = tid; i < 16 * 64; i += 256) Wl[i] = (_Float16)wgt[f0 * 64 + i];
  if (tid < 64) meanl[tid] = meanw[b * 64 + tid];
  for (int i = tid; i < 16 * 256; i += 256) {       // ones-column block rows 64..79
    int row = 64 + (i >> 8), col = i & 255;
    vtl[row * 272 + col] = (row == 64) ? (_Float16)1.0f : (_Float16)0.0f;
  }
  float Mkb = Mk[b];
  __syncthreads();

  int lane = tid & 31, w = tid >> 5, hi = lane >> 4, ln = lane & 15;
  int slocal = 32 * w;                               // wave-private 32-s chunk
  v8f acc[5] = {};                                   // kv accumulators [16f x 80n]
  const float* kb = kk + (size_t)b * 4096 * 64;
  const float* vb = vv + (size_t)b * 4096 * 64;

  for (int it = 0; it < 16; ++it) {
    int sbase = it * 256;
    for (int i = tid; i < 256 * 64; i += 256) {      // centered/scaled k chunk
      int d = i & 63;
      kcl[i] = (_Float16)((kb[sbase * 64 + i] - meanl[d]) * INV_SIGMA);
    }
    for (int i = tid; i < 64 * 256; i += 256) {      // v chunk, transposed
      int d = i & 63, s = i >> 6;
      vtl[d * 272 + s] = (_Float16)vb[(sbase + s) * 64 + d];
    }
    __syncthreads();
    {                                                // 0.5*|kc|^2 per row
      float a2 = 0.f;
      for (int d = 0; d < 64; ++d) { float x = (float)kcl[tid * 64 + d]; a2 += x * x; }
      norml[tid] = 0.5f * a2;
    }
    __syncthreads();

    // proj^T = Wstrip(A, M=f) x kc^T(B, N=s); exp -> phi^T into wave scratch
    for (int st = 0; st < 2; ++st) {
      v8f accP = {};
      int scol = slocal + 16 * st + ln;
      for (int c = 0; c < 2; ++c) {
        v16h a  = ldsA(Wl + ln * 64 + 32 * c, hi);
        v16h bm = ldsB(kcl + scol * 64 + 32 * c, hi);
        accP = wmma_f16(a, bm, accP);
      }
      float nrm = norml[scol] + Mkb;
#pragma unroll
      for (int r = 0; r < 8; ++r) {
        float ph = __expf(accP[r] - nrm) * FINV;
        phiT[(w * 16 + r + 8 * hi) * 32 + 16 * st + ln] = (_Float16)ph;
      }
    }
    asm volatile("s_wait_dscnt 0" ::: "memory");     // wave-private LDS transpose

    // kv += phi^T(A, K=32 s) x [vT | ones](B)
    for (int t = 0; t < 5; ++t) {
      v16h a  = ldsA(phiT + (w * 16 + ln) * 32, hi);
      v16h bm = ldsB(vtl + (16 * t + ln) * 272 + slocal, hi);
      acc[t] = wmma_f16(a, bm, acc[t]);
    }
    __syncthreads();
  }

#pragma unroll
  for (int t = 0; t < 5; ++t)
#pragma unroll
    for (int r = 0; r < 8; ++r)
      kvred[w * 1280 + (r + 8 * hi) * 80 + 16 * t + ln] = acc[t][r];
  __syncthreads();
  for (int i = tid; i < 1280; i += 256) {            // fixed-order cross-wave sum
    int f = i / 80, n = i % 80;
    float s2 = 0.f;
    for (int w2 = 0; w2 < 8; ++w2) s2 += kvred[w2 * 1280 + i];
    int fa = f0 + f;
    if (n < 64)       kvw[((size_t)b * 256 + fa) * 64 + n] = s2;
    else if (n == 64) ksumw[b * 256 + fa] = s2;
  }
}

// ---- K5: fused q projection, softmax-style phi, deno, out GEMM -------------
__global__ __launch_bounds__(256) void orf_query_out(const float* __restrict__ qq,
                                                     const float* __restrict__ wgt,
                                                     const float* __restrict__ kvw,
                                                     const float* __restrict__ ksumw,
                                                     float* __restrict__ outp) {
  extern __shared__ __align__(64) char smem[];
  _Float16* Bbuf  = (_Float16*)smem;             // W [256x64] then kvT [64 x 272]
  _Float16* qtile = (_Float16*)(smem + 34816);   // 128*64 f16   (16384)
  _Float16* phib  = (_Float16*)(smem + 51200);   // 8 x 16 x 256 f16 (65536)
  float*    ksuml = (float*)(smem + 116736);     // 256 f32
  float*    qn    = (float*)(smem + 117760);     // 128 f32
  int b = blockIdx.y, sc = blockIdx.x, tid = threadIdx.x;
  int s0 = sc * 128;

  for (int i = tid; i < 256 * 64; i += 256) Bbuf[i] = (_Float16)wgt[i];
  ksuml[tid] = ksumw[b * 256 + tid];
  const float* qb = qq + ((size_t)b * 4096 + s0) * 64;
  for (int i = tid; i < 128 * 64; i += 256) qtile[i] = (_Float16)(qb[i] * INV_SIGMA);
  __syncthreads();
  if (tid < 128) {
    float a2 = 0.f;
    for (int d = 0; d < 64; ++d) { float x = (float)qtile[tid * 64 + d]; a2 += x * x; }
    qn[tid] = 0.5f * a2;
  }
  __syncthreads();

  int lane = tid & 31, w = tid >> 5, hi = lane >> 4, ln = lane & 15;

  // pass A: q @ W^T, track row maxima (rows r+8*hi live in half 'hi')
  float rmax[8];
#pragma unroll
  for (int r = 0; r < 8; ++r) rmax[r] = -1e30f;
  for (int t = 0; t < 16; ++t) {
    v8f accP = {};
    for (int c = 0; c < 2; ++c) {
      v16h a  = ldsA(qtile + (16 * w + ln) * 64 + 32 * c, hi);
      v16h bm = ldsB(Bbuf + (16 * t + ln) * 64 + 32 * c, hi);
      accP = wmma_f16(a, bm, accP);
    }
#pragma unroll
    for (int r = 0; r < 8; ++r) rmax[r] = fmaxf(rmax[r], accP[r]);
  }
#pragma unroll
  for (int r = 0; r < 8; ++r)
#pragma unroll
    for (int m = 1; m <= 8; m <<= 1) rmax[r] = fmaxf(rmax[r], __shfl_xor(rmax[r], m));

  // pass B: recompute proj, phi_q -> LDS (f16), deno accumulation
  float deno[8] = {};
  for (int t = 0; t < 16; ++t) {
    v8f accP = {};
    for (int c = 0; c < 2; ++c) {
      v16h a  = ldsA(qtile + (16 * w + ln) * 64 + 32 * c, hi);
      v16h bm = ldsB(Bbuf + (16 * t + ln) * 64 + 32 * c, hi);
      accP = wmma_f16(a, bm, accP);
    }
    float ks = ksuml[16 * t + ln];
#pragma unroll
    for (int r = 0; r < 8; ++r) {
      float ph = __expf(accP[r] - qn[16 * w + r + 8 * hi] - rmax[r]) * FINV;
      deno[r] += ph * ks;
      phib[(w * 16 + r + 8 * hi) * 256 + 16 * t + ln] = (_Float16)ph;
    }
  }
  float invd[8];
#pragma unroll
  for (int r = 0; r < 8; ++r) {
    float dsum = deno[r];
#pragma unroll
    for (int m = 1; m <= 8; m <<= 1) dsum += __shfl_xor(dsum, m);
    invd[r] = 1.0f / fmaxf(dsum, 1e-4f);
  }
  __syncthreads();

  // reload Bbuf as kv^T [64 d][272-stride f]
  for (int i = tid; i < 64 * 256; i += 256) {
    int d = i & 63, f = i >> 6;
    Bbuf[d * 272 + f] = (_Float16)kvw[((size_t)b * 256 + f) * 64 + d];
  }
  __syncthreads();

  // out = (phi_q @ kv) * invd
  float* ob = outp + ((size_t)b * 4096 + s0) * 64;
  for (int t = 0; t < 4; ++t) {
    v8f accO = {};
    for (int c = 0; c < 8; ++c) {
      v16h a  = ldsA(phib + (16 * w + ln) * 256 + 32 * c, hi);
      v16h bm = ldsB(Bbuf + (16 * t + ln) * 272 + 32 * c, hi);
      accO = wmma_f16(a, bm, accO);
    }
#pragma unroll
    for (int r = 0; r < 8; ++r)
      ob[(16 * w + r + 8 * hi) * 64 + 16 * t + ln] = accO[r] * invd[r];
  }
}

// ---- host launch ------------------------------------------------------------
extern "C" void kernel_launch(void* const* d_in, const int* in_sizes, int n_in,
                              void* d_out, int out_size, void* d_ws, size_t ws_size,
                              hipStream_t stream) {
  const float* q   = (const float*)d_in[0];
  const float* k   = (const float*)d_in[1];
  const float* v   = (const float*)d_in[2];
  const float* wgt = (const float*)d_in[3];
  float* out = (float*)d_out;

  char* ws = (char*)d_ws;
  float* meanw = (float*)(ws + 0);       // 64*64*4      = 16384
  float* bmax  = (float*)(ws + 16384);   // 64*32*4      = 8192
  float* Mk    = (float*)(ws + 24576);   // 64*4
  float* ksumw = (float*)(ws + 32768);   // 64*256*4     = 65536
  float* kvw   = (float*)(ws + 98304);   // 64*256*64*4  = 4 MiB

  (void)hipFuncSetAttribute((const void*)orf_kproj_max,
                            hipFuncAttributeMaxDynamicSharedMemorySize, 49440);
  (void)hipFuncSetAttribute((const void*)orf_phik_kv,
                            hipFuncAttributeMaxDynamicSharedMemorySize, 128768);
  (void)hipFuncSetAttribute((const void*)orf_query_out,
                            hipFuncAttributeMaxDynamicSharedMemorySize, 118272);

  orf_mean_k<<<64, 256, 0, stream>>>(k, meanw);
  orf_kproj_max<<<dim3(32, 64), 256, 49440, stream>>>(k, wgt, meanw, bmax);
  orf_max_reduce<<<64, 32, 0, stream>>>(bmax, Mk);
  orf_phik_kv<<<dim3(16, 64), 256, 128768, stream>>>(k, v, wgt, meanw, Mk, kvw, ksumw);
  orf_query_out<<<dim3(32, 64), 256, 118272, stream>>>(q, wgt, kvw, ksumw, out);
}
